// QuesWordSentAttention_60206851555566
// MI455X (gfx1250) — matrix operand order repaired and checked by reading, not compiled, not run on hardware
//
#include <hip/hip_runtime.h>
#include <hip/hip_bf16.h>

typedef __bf16 v16bf __attribute__((ext_vector_type(16)));
typedef float  v8f   __attribute__((ext_vector_type(8)));

#define D_DIM   768
#define N_SEQ   512
#define P_NUM   64
#define LQ_NUM  64
#define L_NUM   448
#define ROWS    (P_NUM * N_SEQ)          /* 32768 */
#define T_TOK   (P_NUM * L_NUM)          /* 28672 */
#define ASTR    34                        /* LDS stride for A tile (bf16 elems) */
#define BSTR    36                        /* LDS stride for B tile (bf16 elems) */

union V16U { v16bf v; unsigned u[8]; };

// ---------------------------------------------------------------------------
// 128x128-block-tile WMMA GEMM.  8 waves, each computes a 32x64 sub-tile as
// 2x4 fragments of 16x16 (8 WMMAs per 32-deep k-step).
//   B_TRANS=1: B given row-major [N,K]  (linear-layer weight / k-matrix)
//   B_TRANS=0: B given row-major [K,N]  (v-matrix)
// EPI: 0 = +bias, store bf16 ; 1 = *scale, store f32 ; 2 = relu, store f32
// ---------------------------------------------------------------------------
template<bool A_F32, bool B_F32, bool B_TRANS, int EPI>
__global__ void __launch_bounds__(256) wmma_gemm(
    const void* __restrict__ Ap, long abatch, int lda,
    const void* __restrict__ Bp, long bbatch, int ldb,
    void*       __restrict__ Cp, long cbatch, int ldc,
    const float* __restrict__ bias, float scale, int K)
{
    __shared__ __align__(16) __bf16 As[128 * ASTR];
    __shared__ __align__(16) __bf16 Bs[128 * BSTR];

    const int bz   = blockIdx.z;
    const int row0 = blockIdx.y * 128;
    const int col0 = blockIdx.x * 128;
    const int t    = threadIdx.x;
    const int lane = t & 31;
    const int wid  = t >> 5;
    const int wm   = (wid & 3) * 32;     // wave row base inside tile (0..96)
    const int wn   = (wid >> 2) * 64;    // wave col base inside tile (0 or 64)
    const int lm   = lane & 15;
    const int lhi  = lane >> 4;

    v8f acc[2][4];
    #pragma unroll
    for (int s = 0; s < 2; ++s)
        #pragma unroll
        for (int f = 0; f < 4; ++f) acc[s][f] = (v8f){};

    // global->LDS load assignments (256 threads, 16 elems each)
    const int ar  = t >> 1;          // 0..127  A row
    const int ak  = (t & 1) * 16;    // 0 or 16
    const int bnr = t >> 1;          // 0..127  B row (trans mode, n index)
    const int bkc = (t & 1) * 16;
    const int nbk = t >> 3;          // 0..31   B row (normal mode, k index)
    const int nbn = (t & 7) * 16;    // n start

    for (int kt = 0; kt < K; kt += 32) {
        // ---- A tile [128 x 32] -> As[row][k] (bf16) ----
        if (A_F32) {
            const float* Arow = (const float*)Ap + (long)bz * abatch
                              + (size_t)(row0 + ar) * lda + kt + ak;
            #pragma unroll
            for (int i = 0; i < 16; ++i) As[ar * ASTR + ak + i] = (__bf16)Arow[i];
        } else {
            const __bf16* Arow = (const __bf16*)Ap + (long)bz * abatch
                               + (size_t)(row0 + ar) * lda + kt + ak;
            #pragma unroll
            for (int i = 0; i < 16; ++i) As[ar * ASTR + ak + i] = Arow[i];
        }
        // ---- B tile -> Bs[n][k] (bf16) ----
        if (B_TRANS) {
            if (B_F32) {
                const float* Brow = (const float*)Bp + (long)bz * bbatch
                                  + (size_t)(col0 + bnr) * ldb + kt + bkc;
                #pragma unroll
                for (int i = 0; i < 16; ++i) Bs[bnr * BSTR + bkc + i] = (__bf16)Brow[i];
            } else {
                const __bf16* Brow = (const __bf16*)Bp + (long)bz * bbatch
                                   + (size_t)(col0 + bnr) * ldb + kt + bkc;
                #pragma unroll
                for (int i = 0; i < 16; ++i) Bs[bnr * BSTR + bkc + i] = Brow[i];
            }
        } else {
            const __bf16* Brow = (const __bf16*)Bp + (long)bz * bbatch
                               + (size_t)(kt + nbk) * ldb + col0 + nbn;
            #pragma unroll
            for (int i = 0; i < 16; ++i) Bs[(nbn + i) * BSTR + nbk] = Brow[i];
        }
        __syncthreads();

        // ---- assemble fragments per ISA 7.12.2 (bf16, 16x16x32) ----
        // A: lane m = lane&15 ; K pairs: j<4 -> k=2j+8*lhi ; j>=4 -> k=2j+8+8*lhi
        V16U afr[2];
        #pragma unroll
        for (int s = 0; s < 2; ++s) {
            const unsigned* arow_u = (const unsigned*)&As[(wm + 16 * s + lm) * ASTR];
            #pragma unroll
            for (int j = 0; j < 8; ++j)
                afr[s].u[j] = arow_u[(j < 4 ? j : j + 4) + lhi * 4];
        }
        // B: lane n = lane&15 ; K pairs: j<4 -> k=2j+8*lhi ; j>=4 -> k=2j+8+8*lhi  (Bs stored [n][k])
        V16U bfr[4];
        #pragma unroll
        for (int f = 0; f < 4; ++f) {
            const unsigned* brow_u = (const unsigned*)&Bs[(wn + 16 * f + lm) * BSTR];
            #pragma unroll
            for (int j = 0; j < 8; ++j)
                bfr[f].u[j] = brow_u[(j < 4 ? j : j + 4) + lhi * 4];
        }

        #pragma unroll
        for (int s = 0; s < 2; ++s)
            #pragma unroll
            for (int f = 0; f < 4; ++f)
                acc[s][f] = __builtin_amdgcn_wmma_f32_16x16x32_bf16(
                    false, afr[s].v, false, bfr[f].v, (short)0, acc[s][f],
                    false, false);
        __syncthreads();
    }

    // ---- epilogue: C/D layout -> VGPR r holds (m = r + 8*lhi, n = lane&15) ----
    #pragma unroll
    for (int s = 0; s < 2; ++s) {
        #pragma unroll
        for (int f = 0; f < 4; ++f) {
            const int col = col0 + wn + 16 * f + lm;
            float bv = 0.0f;
            if (EPI == 0) bv = bias[col];
            #pragma unroll
            for (int r = 0; r < 8; ++r) {
                const int row = row0 + wm + 16 * s + r + lhi * 8;
                float xv = acc[s][f][r];
                if (EPI == 0) {
                    __bf16* C = (__bf16*)Cp + (long)bz * cbatch;
                    C[(size_t)row * ldc + col] = (__bf16)(xv + bv);
                } else if (EPI == 1) {
                    float* C = (float*)Cp + (long)bz * cbatch;
                    C[(size_t)row * ldc + col] = xv * scale;
                } else {
                    float* C = (float*)Cp + (long)bz * cbatch;
                    C[(size_t)row * ldc + col] = fmaxf(xv, 0.0f);
                }
            }
        }
    }
}

// ---------------------------------------------------------------------------
// x[p,n,:] = (n<64) ? ques_hidden + ques_share : words * alpha_sent[sent_ids]
// ---------------------------------------------------------------------------
__global__ void __launch_bounds__(256) prep_x_kernel(
    const float* __restrict__ qh, const float* __restrict__ qs,
    const float* __restrict__ wh, const int* __restrict__ sid,
    const float* __restrict__ alpha, float* __restrict__ x)
{
    const int row = blockIdx.x;          // 0..32767
    const int p = row >> 9;
    const int n = row & 511;
    float* xr = x + (size_t)row * D_DIM;
    const int t = threadIdx.x;
    if (n < LQ_NUM) {
        const float* q0 = qh + ((size_t)p * LQ_NUM + n) * D_DIM;
        const float* s0 = qs + (size_t)n * D_DIM;
        #pragma unroll
        for (int i = 0; i < 3; ++i) { int d = t + 256 * i; xr[d] = q0[d] + s0[d]; }
    } else {
        const int tok = p * L_NUM + (n - LQ_NUM);
        const float a = alpha[sid[tok]];
        const float* w0 = wh + (size_t)tok * D_DIM;
        #pragma unroll
        for (int i = 0; i < 3; ++i) { int d = t + 256 * i; xr[d] = w0[d] * a; }
    }
}

// ---------------------------------------------------------------------------
// row softmax over 512 fp32 scores -> bf16 attn
// ---------------------------------------------------------------------------
__global__ void __launch_bounds__(256) softmax_kernel(
    const float* __restrict__ z, __bf16* __restrict__ attn)
{
    __shared__ float red[256];
    const int row = blockIdx.x;
    const int t = threadIdx.x;
    const float* zr = z + (size_t)row * N_SEQ;
    float a0 = zr[t], a1 = zr[t + 256];
    red[t] = fmaxf(a0, a1);
    __syncthreads();
    for (int s = 128; s > 0; s >>= 1) {
        if (t < s) red[t] = fmaxf(red[t], red[t + s]);
        __syncthreads();
    }
    const float mx = red[0];
    __syncthreads();
    float e0 = __expf(a0 - mx), e1 = __expf(a1 - mx);
    red[t] = e0 + e1;
    __syncthreads();
    for (int s = 128; s > 0; s >>= 1) {
        if (t < s) red[t] += red[t + s];
        __syncthreads();
    }
    const float inv = 1.0f / red[0];
    __bf16* ar = attn + (size_t)row * N_SEQ;
    ar[t]       = (__bf16)(e0 * inv);
    ar[t + 256] = (__bf16)(e1 * inv);
}

// ---------------------------------------------------------------------------
// x = LayerNorm(x + a) ; a already ReLU'ed
// ---------------------------------------------------------------------------
__global__ void __launch_bounds__(256) add_ln_kernel(
    float* __restrict__ x, const float* __restrict__ a,
    const float* __restrict__ g, const float* __restrict__ b)
{
    __shared__ float red[256];
    const int row = blockIdx.x;
    const int t = threadIdx.x;
    float* xr = x + (size_t)row * D_DIM;
    const float* ar = a + (size_t)row * D_DIM;
    float v0 = xr[t]       + ar[t];
    float v1 = xr[t + 256] + ar[t + 256];
    float v2 = xr[t + 512] + ar[t + 512];
    red[t] = v0 + v1 + v2;
    __syncthreads();
    for (int s = 128; s > 0; s >>= 1) {
        if (t < s) red[t] += red[t + s];
        __syncthreads();
    }
    const float mu = red[0] * (1.0f / (float)D_DIM);
    __syncthreads();
    const float d0 = v0 - mu, d1 = v1 - mu, d2 = v2 - mu;
    red[t] = d0 * d0 + d1 * d1 + d2 * d2;
    __syncthreads();
    for (int s = 128; s > 0; s >>= 1) {
        if (t < s) red[t] += red[t + s];
        __syncthreads();
    }
    const float rstd = rsqrtf(red[0] * (1.0f / (float)D_DIM) + 1e-5f);
    xr[t]       = d0 * rstd * g[t]       + b[t];
    xr[t + 256] = d1 * rstd * g[t + 256] + b[t + 256];
    xr[t + 512] = d2 * rstd * g[t + 512] + b[t + 512];
}

// ---------------------------------------------------------------------------
// outputs: [0 .. 64*64*768)      ques_update = ques_ori + x[:, :64]
//          [64*64*768 .. end)    words_update = words + x[:, 64:]
// ---------------------------------------------------------------------------
__global__ void __launch_bounds__(256) final_kernel(
    const float* __restrict__ qh, const float* __restrict__ qs,
    const float* __restrict__ wh, const float* __restrict__ x,
    float* __restrict__ out)
{
    const size_t WOFF = (size_t)P_NUM * LQ_NUM * D_DIM;  // 3145728
    const int row = blockIdx.x;
    const int p = row >> 9;
    const int n = row & 511;
    const int t = threadIdx.x;
    const float* xr = x + (size_t)row * D_DIM;
    if (n < LQ_NUM) {
        const size_t o = ((size_t)p * LQ_NUM + n) * D_DIM;
        #pragma unroll
        for (int i = 0; i < 3; ++i) {
            int d = t + 256 * i;
            out[o + d] = qh[o + d] + qs[(size_t)n * D_DIM + d] + xr[d];
        }
    } else {
        const size_t tok = (size_t)p * L_NUM + (n - LQ_NUM);
        #pragma unroll
        for (int i = 0; i < 3; ++i) {
            int d = t + 256 * i;
            out[WOFF + tok * D_DIM + d] = wh[tok * D_DIM + d] + xr[d];
        }
    }
}

// ---------------------------------------------------------------------------
extern "C" void kernel_launch(void* const* d_in, const int* in_sizes, int n_in,
                              void* d_out, int out_size, void* d_ws, size_t ws_size,
                              hipStream_t stream)
{
    (void)in_sizes; (void)n_in; (void)out_size; (void)ws_size;
    const float* ques_hidden = (const float*)d_in[0];
    const float* ques_share  = (const float*)d_in[1];
    const float* words       = (const float*)d_in[2];
    const int*   sent_ids    = (const int*)d_in[3];
    const float* alpha_sent  = (const float*)d_in[4];
    const float* Wq = (const float*)d_in[6];
    const float* bq = (const float*)d_in[7];
    const float* Wk = (const float*)d_in[8];
    const float* bk = (const float*)d_in[9];
    const float* Wv = (const float*)d_in[10];
    const float* bv = (const float*)d_in[11];
    const float* ln_g = (const float*)d_in[12];
    const float* ln_b = (const float*)d_in[13];

    // workspace layout (bytes)
    char* ws = (char*)d_ws;
    float*  x    = (float*) (ws);                       // 96 MB  [32768 x 768] f32
    __bf16* qb   = (__bf16*)(ws + 100663296UL);         // 48 MB
    __bf16* kb   = (__bf16*)(ws + 150994944UL);         // 48 MB
    __bf16* vb   = (__bf16*)(ws + 201326592UL);         // 48 MB
    float*  sc   = (float*) (ws + 251658240UL);         // 64 MB  [64 x 512 x 512] f32
    __bf16* attn = (__bf16*)(ws + 318767104UL);         // 32 MB
    float*  abuf = (float*) (ws + 100663296UL);         // 96 MB, aliases q|k (dead by then)

    const float scale = 1.0f / sqrtf((float)D_DIM);
    const long QKV_BS = (long)N_SEQ * D_DIM;   // 393216
    const long SC_BS  = (long)N_SEQ * N_SEQ;   // 262144

    prep_x_kernel<<<ROWS, 256, 0, stream>>>(ques_hidden, ques_share, words,
                                            sent_ids, alpha_sent, x);

    for (int i = 0; i < 2; ++i) {
        const float* Wq_i = Wq + (size_t)i * D_DIM * D_DIM;
        const float* Wk_i = Wk + (size_t)i * D_DIM * D_DIM;
        const float* Wv_i = Wv + (size_t)i * D_DIM * D_DIM;

        // q/k/v = x @ W^T + b        (M=32768, N=768, K=768)
        dim3 gqkv(D_DIM / 128, ROWS / 128, 1);           // 6 x 256
        wmma_gemm<true, true, true, 0><<<gqkv, 256, 0, stream>>>(
            x, 0, D_DIM, Wq_i, 0, D_DIM, qb, 0, D_DIM, bq + i * D_DIM, 0.0f, D_DIM);
        wmma_gemm<true, true, true, 0><<<gqkv, 256, 0, stream>>>(
            x, 0, D_DIM, Wk_i, 0, D_DIM, kb, 0, D_DIM, bk + i * D_DIM, 0.0f, D_DIM);
        wmma_gemm<true, true, true, 0><<<gqkv, 256, 0, stream>>>(
            x, 0, D_DIM, Wv_i, 0, D_DIM, vb, 0, D_DIM, bv + i * D_DIM, 0.0f, D_DIM);

        // scores = scale * q @ k^T   (batched, M=N=512, K=768)
        dim3 gsc(N_SEQ / 128, N_SEQ / 128, P_NUM);       // 4 x 4 x 64
        wmma_gemm<false, false, true, 1><<<gsc, 256, 0, stream>>>(
            qb, QKV_BS, D_DIM, kb, QKV_BS, D_DIM, sc, SC_BS, N_SEQ,
            nullptr, scale, D_DIM);

        softmax_kernel<<<ROWS, 256, 0, stream>>>(sc, attn);

        // a = relu(attn @ v)         (batched, M=512, N=768, K=512)
        dim3 gav(D_DIM / 128, N_SEQ / 128, P_NUM);       // 6 x 4 x 64
        wmma_gemm<false, false, false, 2><<<gav, 256, 0, stream>>>(
            attn, SC_BS, N_SEQ, vb, QKV_BS, D_DIM, abuf, QKV_BS, D_DIM,
            nullptr, 0.0f, N_SEQ);

        add_ln_kernel<<<ROWS, 256, 0, stream>>>(x, abuf,
                                                ln_g + i * D_DIM, ln_b + i * D_DIM);
    }

    final_kernel<<<ROWS, 256, 0, stream>>>(ques_hidden, ques_share, words, x,
                                           (float*)d_out);
}
